// GATScorer_64707977281658
// MI455X (gfx1250) — matrix operand order repaired and checked by reading, not compiled, not run on hardware
//
#include <hip/hip_runtime.h>
#include <hip/hip_fp16.h>

#define N_NODES 20000
#define N_EDGES 320000
#define N_CAND  200000
#define IN_DIM  7
#define F1      256    // H1*C1
#define H1_     4
#define C1_     64
#define F2      32     // H2*C2
#define KP1     32     // K for conv1 GEMM, zero-padded from 7
#define E_TOT   (N_EDGES + N_NODES)   // self-loops appended -> 340000
#define NEG_SLOPE 0.2f

typedef __attribute__((ext_vector_type(16))) _Float16 v16h;
typedef __attribute__((ext_vector_type(8)))  _Float16 v8h;
typedef __attribute__((ext_vector_type(8)))  float    v8f;

// ---------- monotonic float<->uint encoding for atomic max ----------
__device__ __forceinline__ unsigned fenc(float f) {
  unsigned u = __float_as_uint(f);
  return (u & 0x80000000u) ? ~u : (u | 0x80000000u);
}
__device__ __forceinline__ float fdec(unsigned u) {
  return (u & 0x80000000u) ? __uint_as_float(u & 0x7FFFFFFFu) : __uint_as_float(~u);
}

__device__ __forceinline__ void edge_nodes(int e, const long long* __restrict__ ei,
                                           int& s, int& d) {
  if (e < N_EDGES) { s = (int)ei[e]; d = (int)ei[N_EDGES + e]; }
  else             { s = e - N_EDGES; d = s; }   // self loop
}

__device__ __forceinline__ v16h cat8(v8h lo, v8h hi8) {
  return __builtin_shufflevector(lo, hi8, 0, 1, 2, 3, 4, 5, 6, 7,
                                          8, 9, 10, 11, 12, 13, 14, 15);
}

// ---------------- operand packing (f16, WMMA-friendly layouts) ------------------
// xh[n][k]  : K-padded f16 copy of x            (k<7 real, else 0)
__global__ void prep_xh(const float* __restrict__ x, _Float16* __restrict__ xh) {
  int i = blockIdx.x * blockDim.x + threadIdx.x;
  if (i >= N_NODES * KP1) return;
  int n = i >> 5, k = i & 31;
  xh[i] = (k < IN_DIM) ? (_Float16)x[n * IN_DIM + k] : (_Float16)0.0f;
}
// w1h[col][k] : transposed K-padded f16 W1  (B operand contiguous in K)
__global__ void prep_w1h(const float* __restrict__ W1, _Float16* __restrict__ w1h) {
  int i = blockIdx.x * blockDim.x + threadIdx.x;
  if (i >= F1 * KP1) return;
  int col = i >> 5, k = i & 31;
  w1h[i] = (k < IN_DIM) ? (_Float16)W1[k * F1 + col] : (_Float16)0.0f;
}
// w2h[col][k] : transposed f16 W2 (32 cols x 256 K)
__global__ void prep_w2h(const float* __restrict__ W2, _Float16* __restrict__ w2h) {
  int i = blockIdx.x * blockDim.x + threadIdx.x;
  if (i >= F2 * F1) return;
  int col = i >> 8, k = i & 255;
  w2h[i] = (_Float16)W2[k * F2 + col];
}

// ---------------- GEMM1: h1pre[N,256] = x[N,7] @ W1[7,256] (one WMMA / tile) -----
__global__ void gemm1_wmma(const _Float16* __restrict__ xh, const _Float16* __restrict__ w1h,
                           float* __restrict__ h1pre) {
  const int m0 = blockIdx.x * 16;
  const int n0 = blockIdx.y * 16;
  const int lane = threadIdx.x;
  const int rlo  = lane & 15;
  const bool hi  = lane >= 16;
  const int koff = hi ? 8 : 0;    // A: elems 0..7 -> K koff.., elems 8..15 -> K koff+16..
  const int kb0  = hi ? 8 : 0;    // B: elems 0..7 -> K kb0.., elems 8..15 -> K kb0+16..
  const _Float16* ap = xh  + (m0 + rlo) * KP1 + koff;
  const _Float16* bp = w1h + (n0 + rlo) * KP1 + kb0;
  v16h a = cat8(*(const v8h*)ap, *(const v8h*)(ap + 16));
  v16h b = cat8(*(const v8h*)bp, *(const v8h*)(bp + 16));
  v8f c = {};
  c = __builtin_amdgcn_wmma_f32_16x16x32_f16(false, a, false, b, (short)0, c, false, false);
#pragma unroll
  for (int r = 0; r < 8; ++r) {
    int rr = m0 + r + (hi ? 8 : 0);
    h1pre[rr * F1 + n0 + rlo] = c[r];
  }
}

// ---------------- GEMM2: h2pre[N,32] = h1[N,256] @ W2[256,32] -------------------
__global__ void gemm2_wmma(const _Float16* __restrict__ h1h, const _Float16* __restrict__ w2h,
                           float* __restrict__ h2pre) {
  const int m0 = blockIdx.x * 16;
  const int n0 = blockIdx.y * 16;
  const int lane = threadIdx.x;
  const int rlo  = lane & 15;
  const bool hi  = lane >= 16;
  const int koff = hi ? 8 : 0;
  const int kb0  = hi ? 8 : 0;
  const _Float16* ap = h1h + (m0 + rlo) * F1 + koff;
  const _Float16* bp = w2h + (n0 + rlo) * F1 + kb0;
  v8f c = {};
#pragma unroll
  for (int k0 = 0; k0 < F1; k0 += 32) {
    v16h a = cat8(*(const v8h*)(ap + k0), *(const v8h*)(ap + k0 + 16));
    v16h b = cat8(*(const v8h*)(bp + k0), *(const v8h*)(bp + k0 + 16));
    c = __builtin_amdgcn_wmma_f32_16x16x32_f16(false, a, false, b, (short)0, c, false, false);
  }
#pragma unroll
  for (int r = 0; r < 8; ++r) {
    int rr = m0 + r + (hi ? 8 : 0);
    h2pre[rr * F2 + n0 + rlo] = c[r];
  }
}

// ---------------- attention scalars: alpha_s/alpha_d ----------------------------
__global__ void alpha1_k(const float* __restrict__ h1pre, const float* __restrict__ a_src,
                         const float* __restrict__ a_dst, float* __restrict__ as,
                         float* __restrict__ ad) {
  int i = blockIdx.x * blockDim.x + threadIdx.x;
  if (i >= N_NODES * H1_) return;
  int n = i >> 2, h = i & 3;
  const float4* hp  = (const float4*)(h1pre + n * F1 + h * C1_);
  const float4* asv = (const float4*)(a_src + h * C1_);
  const float4* adv = (const float4*)(a_dst + h * C1_);
  float ss = 0.f, sd = 0.f;
#pragma unroll
  for (int q = 0; q < C1_ / 4; ++q) {
    float4 v = hp[q], sa = asv[q], da = adv[q];
    ss += v.x * sa.x + v.y * sa.y + v.z * sa.z + v.w * sa.w;
    sd += v.x * da.x + v.y * da.y + v.z * da.z + v.w * da.w;
  }
  as[i] = ss; ad[i] = sd;
}

__global__ void alpha2_k(const float* __restrict__ h2pre, const float* __restrict__ a_src,
                         const float* __restrict__ a_dst, float* __restrict__ as,
                         float* __restrict__ ad) {
  int n = blockIdx.x * blockDim.x + threadIdx.x;
  if (n >= N_NODES) return;
  const float4* hp = (const float4*)(h2pre + n * F2);
  float ss = 0.f, sd = 0.f;
#pragma unroll
  for (int q = 0; q < F2 / 4; ++q) {
    float4 v = hp[q];
    float4 sa = ((const float4*)a_src)[q];
    float4 da = ((const float4*)a_dst)[q];
    ss += v.x * sa.x + v.y * sa.y + v.z * sa.z + v.w * sa.w;
    sd += v.x * da.x + v.y * da.y + v.z * da.z + v.w * da.w;
  }
  as[n] = ss; ad[n] = sd;
}

// ---------------- edge pass 1: leaky-relu logit + segment max -------------------
template <int H>
__global__ void logits_k(const long long* __restrict__ ei, const float* __restrict__ as,
                         const float* __restrict__ ad, float* __restrict__ ebuf,
                         unsigned* __restrict__ menc) {
  int e = blockIdx.x * blockDim.x + threadIdx.x;
  if (e >= E_TOT) return;
  int s, d; edge_nodes(e, ei, s, d);
#pragma unroll
  for (int h = 0; h < H; ++h) {
    float v = as[s * H + h] + ad[d * H + h];
    v = (v >= 0.f) ? v : NEG_SLOPE * v;
    ebuf[e * H + h] = v;
    atomicMax(&menc[d * H + h], fenc(v));
  }
}

// ---------------- edge pass 2: exp(e - m) + segment sum -------------------------
template <int H>
__global__ void sumexp_k(const long long* __restrict__ ei, const unsigned* __restrict__ menc,
                         float* __restrict__ ebuf, float* __restrict__ z) {
  int e = blockIdx.x * blockDim.x + threadIdx.x;
  if (e >= E_TOT) return;
  int s, d; edge_nodes(e, ei, s, d);
  (void)s;
#pragma unroll
  for (int h = 0; h < H; ++h) {
    float v = expf(ebuf[e * H + h] - fdec(menc[d * H + h]));
    ebuf[e * H + h] = v;
    atomicAdd(&z[d * H + h], v);
  }
}

// ---------------- edge pass 3: weighted message scatter -------------------------
// conv1: one 256-thread block per edge; thread t = channel, head = t/64
__global__ void scatter1_k(const long long* __restrict__ ei, const float* __restrict__ h1pre,
                           const float* __restrict__ ebuf, const float* __restrict__ z,
                           float* __restrict__ out1) {
  int e = blockIdx.x;
  int t = threadIdx.x;
  int s, d; edge_nodes(e, ei, s, d);
  int h = t >> 6;
  float coef = ebuf[e * H1_ + h] / (z[d * H1_ + h] + 1e-16f);
  atomicAdd(&out1[d * F1 + t], h1pre[s * F1 + t] * coef);
}

// conv2: 8 edges per 256-thread block, 32 channels each
__global__ void scatter2_k(const long long* __restrict__ ei, const float* __restrict__ h2pre,
                           const float* __restrict__ ebuf, const float* __restrict__ z,
                           float* __restrict__ out2) {
  int t = threadIdx.x;
  int e = blockIdx.x * 8 + (t >> 5);
  int c = t & 31;
  if (e >= E_TOT) return;
  int s, d; edge_nodes(e, ei, s, d);
  float coef = ebuf[e] / (z[d] + 1e-16f);
  atomicAdd(&out2[d * F2 + c], h2pre[s * F2 + c] * coef);
}

// ---------------- ELU + bias, emitting f16 copy for the WMMA GEMM2 --------------
__global__ void elu_bias1_k(const float* __restrict__ out1, const float* __restrict__ b1,
                            _Float16* __restrict__ h1h) {
  int i = blockIdx.x * blockDim.x + threadIdx.x;
  if (i >= N_NODES * F1) return;
  float v = out1[i] + b1[i & (F1 - 1)];
  v = (v > 0.f) ? v : expm1f(v);   // ELU, alpha=1
  h1h[i] = (_Float16)v;            // GEMM2 consumes f16 anyway
}

__global__ void bias2_k(float* __restrict__ out2, const float* __restrict__ b2) {
  int i = blockIdx.x * blockDim.x + threadIdx.x;
  if (i >= N_NODES * F2) return;
  out2[i] += b2[i & (F2 - 1)];
}

// ---------------- candidate link scoring ----------------------------------------
__global__ void score_k(const long long* __restrict__ cand, const float* __restrict__ h2,
                        const float* __restrict__ Ws, const float* __restrict__ bs,
                        float* __restrict__ out) {
  int i = blockIdx.x * blockDim.x + threadIdx.x;
  if (i >= N_CAND) return;
  int a = (int)cand[2 * i + 0];
  int b = (int)cand[2 * i + 1];
  const float4* ha = (const float4*)(h2 + a * F2);
  const float4* hb = (const float4*)(h2 + b * F2);
  const float4* w0 = (const float4*)Ws;         // Ws[0..31]
  const float4* w1 = (const float4*)(Ws + F2);  // Ws[32..63]
  float s = bs[0];
#pragma unroll
  for (int q = 0; q < F2 / 4; ++q) {
    float4 va = ha[q], vb = hb[q], wa = w0[q], wb = w1[q];
    s += va.x * wa.x + va.y * wa.y + va.z * wa.z + va.w * wa.w;
    s += vb.x * wb.x + vb.y * wb.y + vb.z * wb.z + vb.w * wb.w;
  }
  out[i] = 1.f / (1.f + expf(-s));
}

// =================================================================================
extern "C" void kernel_launch(void* const* d_in, const int* in_sizes, int n_in,
                              void* d_out, int out_size, void* d_ws, size_t ws_size,
                              hipStream_t stream) {
  (void)in_sizes; (void)n_in; (void)out_size; (void)ws_size;
  const float*     x      = (const float*)d_in[0];
  const long long* ei     = (const long long*)d_in[1];   // int64 [2, E]
  const long long* cand   = (const long long*)d_in[2];   // int64 [NC, 2]
  const float*     W1     = (const float*)d_in[3];
  const float*     a_src1 = (const float*)d_in[4];
  const float*     a_dst1 = (const float*)d_in[5];
  const float*     b1     = (const float*)d_in[6];
  const float*     W2     = (const float*)d_in[7];
  const float*     a_src2 = (const float*)d_in[8];
  const float*     a_dst2 = (const float*)d_in[9];
  const float*     b2     = (const float*)d_in[10];
  const float*     Ws     = (const float*)d_in[11];
  const float*     bs     = (const float*)d_in[12];
  float* out = (float*)d_out;

  // ---- carve workspace (~60 MB) ----
  char* base = (char*)d_ws;
  size_t off = 0;
  auto carve = [&](size_t bytes) { void* p = base + off; off = (off + bytes + 255) & ~(size_t)255; return p; };
  float*     h1pre = (float*)    carve((size_t)N_NODES * F1 * 4);
  float*     out1  = (float*)    carve((size_t)N_NODES * F1 * 4);
  float*     as1   = (float*)    carve((size_t)N_NODES * H1_ * 4);
  float*     ad1   = (float*)    carve((size_t)N_NODES * H1_ * 4);
  unsigned*  m1    = (unsigned*) carve((size_t)N_NODES * H1_ * 4);
  float*     z1    = (float*)    carve((size_t)N_NODES * H1_ * 4);
  float*     e1    = (float*)    carve((size_t)E_TOT * H1_ * 4);
  float*     h2pre = (float*)    carve((size_t)N_NODES * F2 * 4);
  float*     out2  = (float*)    carve((size_t)N_NODES * F2 * 4);
  float*     as2   = (float*)    carve((size_t)N_NODES * 4);
  float*     ad2   = (float*)    carve((size_t)N_NODES * 4);
  unsigned*  m2    = (unsigned*) carve((size_t)N_NODES * 4);
  float*     z2    = (float*)    carve((size_t)N_NODES * 4);
  float*     e2    = (float*)    carve((size_t)E_TOT * 4);
  _Float16*  xh    = (_Float16*) carve((size_t)N_NODES * KP1 * 2);
  _Float16*  w1h   = (_Float16*) carve((size_t)F1 * KP1 * 2);
  _Float16*  w2h   = (_Float16*) carve((size_t)F2 * F1 * 2);
  _Float16*  h1h   = (_Float16*) carve((size_t)N_NODES * F1 * 2);

  // ---- zero accumulators (graph-capture-safe async memsets) ----
  hipMemsetAsync(out1, 0, (size_t)N_NODES * F1 * 4, stream);
  hipMemsetAsync(m1,   0, (size_t)N_NODES * H1_ * 4, stream);  // fenc lower bound
  hipMemsetAsync(z1,   0, (size_t)N_NODES * H1_ * 4, stream);
  hipMemsetAsync(out2, 0, (size_t)N_NODES * F2 * 4, stream);
  hipMemsetAsync(m2,   0, (size_t)N_NODES * 4, stream);
  hipMemsetAsync(z2,   0, (size_t)N_NODES * 4, stream);

  const int EB = (E_TOT + 255) / 256;

  // ---- pack WMMA operands ----
  prep_xh <<<(N_NODES * KP1 + 255) / 256, 256, 0, stream>>>(x, xh);
  prep_w1h<<<(F1 * KP1 + 255) / 256, 256, 0, stream>>>(W1, w1h);
  prep_w2h<<<(F2 * F1 + 255) / 256, 256, 0, stream>>>(W2, w2h);

  // ---- conv1 ----
  gemm1_wmma<<<dim3(N_NODES / 16, F1 / 16), dim3(32), 0, stream>>>(xh, w1h, h1pre);
  alpha1_k<<<(N_NODES * H1_ + 255) / 256, 256, 0, stream>>>(h1pre, a_src1, a_dst1, as1, ad1);
  logits_k<H1_><<<EB, 256, 0, stream>>>(ei, as1, ad1, e1, m1);
  sumexp_k<H1_><<<EB, 256, 0, stream>>>(ei, m1, e1, z1);
  scatter1_k<<<E_TOT, 256, 0, stream>>>(ei, h1pre, e1, z1, out1);
  elu_bias1_k<<<(N_NODES * F1 + 255) / 256, 256, 0, stream>>>(out1, b1, h1h);

  // ---- conv2 ----
  gemm2_wmma<<<dim3(N_NODES / 16, F2 / 16), dim3(32), 0, stream>>>(h1h, w2h, h2pre);
  alpha2_k<<<(N_NODES + 255) / 256, 256, 0, stream>>>(h2pre, a_src2, a_dst2, as2, ad2);
  logits_k<1><<<EB, 256, 0, stream>>>(ei, as2, ad2, e2, m2);
  sumexp_k<1><<<EB, 256, 0, stream>>>(ei, m2, e2, z2);
  scatter2_k<<<(E_TOT + 7) / 8, 256, 0, stream>>>(ei, h2pre, e2, z2, out2);
  bias2_k<<<(N_NODES * F2 + 255) / 256, 256, 0, stream>>>(out2, b2);

  // ---- candidate scoring ----
  score_k<<<(N_CAND + 255) / 256, 256, 0, stream>>>(cand, out2, Ws, bs, out);
}
